// Model_1417339208379
// MI455X (gfx1250) — matrix operand order repaired and checked by reading, not compiled, not run on hardware
//
#include <hip/hip_runtime.h>
#include <hip/hip_bf16.h>
#include <math.h>

// ---------------- model dims ----------------
#define BB     16
#define LL     256
#define TT     1024
#define DMODEL 512
#define NHEAD  8
#define DHEAD  64
#define FFDIM  2048
#define NLAYER 6
#define NSYM   256
#define NMEL   80
#define DDUR   256

typedef __attribute__((ext_vector_type(16))) _Float16 v16h;
typedef __attribute__((ext_vector_type(8)))  _Float16 h8;
typedef __attribute__((ext_vector_type(8)))  float    v8f;

union V16U { v16h v; h8 h[2]; };

// ---- CDNA5 async global->LDS copy (16B per lane), tracked by ASYNCcnt ----
__device__ __forceinline__ void async_ld16(const void* gptr, void* lptr) {
  // low 32 bits of a generic pointer to LDS == group-segment byte offset
  unsigned loff = (unsigned)(size_t)lptr;
  asm volatile("global_load_async_to_lds_b128 %0, %1, off"
               :: "v"(loff), "v"(gptr) : "memory");
}
__device__ __forceinline__ void async_wait0() {
  asm volatile("s_wait_asynccnt 0" ::: "memory");
}

__device__ __forceinline__ float pe_val(int pos, int c, int dmodel) {
  int i2 = c & ~1;
  float f = __expf(-(float)i2 * (__logf(10000.0f) / (float)dmodel));
  float a = (float)pos * f;
  return (c & 1) ? __cosf(a) : __sinf(a);
}

// ---------------- embedding + positional encoding (f32 + f16 shadow) ----------------
__global__ __launch_bounds__(256)
void k_embed(const int* __restrict__ text, const float* __restrict__ emb,
             const float* __restrict__ alpha1, float* __restrict__ x,
             _Float16* __restrict__ xh) {
  int gid = blockIdx.x * 256 + threadIdx.x;            // over B*L*D
  int c   = gid % DMODEL;
  int row = gid / DMODEL;                              // b*L + l
  int l   = row % LL;
  float v = emb[(size_t)text[row] * DMODEL + c] + alpha1[0] * pe_val(l, c, DMODEL);
  x[gid]  = v;
  xh[gid] = (_Float16)v;
}

// ---------------- fp32 -> f16 weight conversion ----------------
__global__ __launch_bounds__(256)
void k_wcvt(const float* __restrict__ s, _Float16* __restrict__ d, int n) {
  int i = blockIdx.x * 256 + threadIdx.x;
  if (i < n) d[i] = (_Float16)s[i];
}

// ---------------- WMMA GEMM: out = act(A[M,K]h * W[N,K]h^T + bias [+ R]) ----------------
// 256 thr = 8 waves; block tile 128x128; wave tile 64x32; K-step 32.
// A/W tiles staged with global_load_async_to_lds_b128.
template<bool RELU, bool RES, bool TRANS>
__global__ __launch_bounds__(256)
void k_gemm(const _Float16* __restrict__ A, const _Float16* __restrict__ W,
            const float* __restrict__ bias, const float* __restrict__ R,
            float* __restrict__ Y, _Float16* __restrict__ Yh,
            int M, int N, int K, int Sdim) {
  __shared__ _Float16 sA[128 * 40];
  __shared__ _Float16 sW[128 * 40];
  const int tid  = threadIdx.x;
  const int lane = tid & 31;
  const int wave = tid >> 5;
  const int wm   = (wave >> 2) * 64;
  const int wn   = (wave & 3) * 32;
  const int bM   = blockIdx.y * 128;
  const int bN   = blockIdx.x * 128;

  v8f acc[4][2];
  for (int i = 0; i < 4; i++)
    for (int j = 0; j < 2; j++)
      for (int e = 0; e < 8; e++) acc[i][j][e] = 0.0f;

  // zero-fill W-tile rows that fall outside N (done once; async never rewrites them)
  #pragma unroll
  for (int u = 0; u < 2; u++) {
    int c = tid * 2 + u, row = c >> 2, seg = c & 3;
    if (bN + row >= N) {
      _Float16* d = sW + row * 40 + seg * 8;
      #pragma unroll
      for (int j = 0; j < 8; j++) d[j] = (_Float16)0.0f;
    }
  }

  for (int k0 = 0; k0 < K; k0 += 32) {
    __syncthreads();                       // previous iteration's reads complete
    #pragma unroll
    for (int u = 0; u < 2; u++) {          // A tile: 128 rows x 32 halfs (M%128==0)
      int c = tid * 2 + u, row = c >> 2, seg = c & 3;
      async_ld16(A + (size_t)(bM + row) * K + k0 + seg * 8, sA + row * 40 + seg * 8);
    }
    #pragma unroll
    for (int u = 0; u < 2; u++) {          // W tile (guard N)
      int c = tid * 2 + u, row = c >> 2, seg = c & 3;
      if (bN + row < N)
        async_ld16(W + (size_t)(bN + row) * K + k0 + seg * 8, sW + row * 40 + seg * 8);
    }
    if (k0 + 32 < K)
      __builtin_prefetch(A + (size_t)(bM + (tid >> 2)) * K + k0 + 32, 0, 0);
    async_wait0();
    __syncthreads();

    const int hi   = lane >> 4;
    const int lrow = lane & 15;
    V16U afr[4], bfr[2];
    #pragma unroll
    for (int mt = 0; mt < 4; mt++) {
      // A 16x32 fragment: lanes<16: K {0-7,16-23}; lanes>=16: K {8-15,24-31}
      const _Float16* p = sA + (wm + mt * 16 + lrow) * 40 + hi * 8;
      afr[mt].h[0] = *(const h8*)p;
      afr[mt].h[1] = *(const h8*)(p + 16);
    }
    #pragma unroll
    for (int nt = 0; nt < 2; nt++) {
      // B 32x16 fragment: lanes<16: K {0-7,16-23}; lanes>=16: K {8-15,24-31} (col = lane%16)
      const _Float16* p = sW + (wn + nt * 16 + lrow) * 40 + hi * 8;
      bfr[nt].h[0] = *(const h8*)p;
      bfr[nt].h[1] = *(const h8*)(p + 16);
    }
    #pragma unroll
    for (int mt = 0; mt < 4; mt++)
      #pragma unroll
      for (int nt = 0; nt < 2; nt++)
        acc[mt][nt] = __builtin_amdgcn_wmma_f32_16x16x32_f16(
            false, afr[mt].v, false, bfr[nt].v, (short)0, acc[mt][nt], false, false);
  }

  // store: C layout -> lane n, reg r: row = r + 8*(n>=16), col = n%16
  const int hi   = lane >> 4;
  const int lcol = lane & 15;
  #pragma unroll
  for (int mt = 0; mt < 4; mt++) {
    #pragma unroll
    for (int nt = 0; nt < 2; nt++) {
      int col = bN + wn + nt * 16 + lcol;
      if (col >= N) continue;
      float bv = bias ? bias[col] : 0.0f;
      #pragma unroll
      for (int r = 0; r < 8; r++) {
        int row = bM + wm + mt * 16 + hi * 8 + r;
        float v = acc[mt][nt][r] + bv;
        if (RES)  v += R[(size_t)row * N + col];
        if (RELU) v = fmaxf(v, 0.0f);
        if (Y) {
          if (TRANS) {
            int bb = row / Sdim, t = row % Sdim;
            Y[((size_t)bb * N + col) * Sdim + t] = v;
          } else {
            Y[(size_t)row * N + col] = v;
          }
        }
        if (Yh) Yh[(size_t)row * N + col] = (_Float16)v;
      }
    }
  }
}

// ---------------- LayerNorm (one wave per row), optional f16 shadow ----------------
__global__ __launch_bounds__(256)
void k_ln(const float* __restrict__ X, const float* __restrict__ g,
          const float* __restrict__ b, float* __restrict__ Y,
          _Float16* __restrict__ Yh, int M, int Dp) {
  int lane = threadIdx.x & 31;
  int row  = blockIdx.x * 8 + (threadIdx.x >> 5);
  if (row >= M) return;
  const float* x = X + (size_t)row * Dp;
  float s = 0.f, s2 = 0.f;
  for (int c = lane; c < Dp; c += 32) { float v = x[c]; s += v; s2 += v * v; }
  for (int o = 1; o < 32; o <<= 1) { s += __shfl_xor(s, o, 32); s2 += __shfl_xor(s2, o, 32); }
  float mean = s / Dp;
  float var  = s2 / Dp - mean * mean;
  float inv  = rsqrtf(var + 1e-5f);
  float* y = Y + (size_t)row * Dp;
  for (int c = lane; c < Dp; c += 32) {
    float v = (x[c] - mean) * inv * g[c] + b[c];
    y[c] = v;
    if (Yh) Yh[(size_t)row * Dp + c] = (_Float16)v;
  }
}

// ---------------- flash attention (WMMA): 4 waves, 16 q-rows per wave ----------------
// qkv is f16 [B,S,3*D]; output f16 [B,S,D]. Row sums via P x ones WMMA.
__global__ __launch_bounds__(128)
void k_attn(const _Float16* __restrict__ qkv, const int* __restrict__ lens,
            _Float16* __restrict__ O, int S) {
  __shared__ _Float16 sK[32 * 72];     // [key][d] f16 (async-staged)
  __shared__ _Float16 sVt[64 * 40];    // [d][key] f16 (transposed)
  __shared__ _Float16 sP[4][16 * 40];  // per-wave P bounce (16x32)
  int bh = blockIdx.x;
  int b = bh / NHEAD, h = bh % NHEAD;
  int tid = threadIdx.x, lane = tid & 31, wave = tid >> 5;
  int qbase = blockIdx.y * 64 + wave * 16;
  int hi = lane >> 4, lrow = lane & 15;
  int seqlen = lens[b];
  const size_t rstride = 3 * DMODEL;
  const _Float16* base = qkv + (size_t)b * S * rstride;

  // Q fragments (vector loads from f16, scale applied to scores later)
  V16U aq0, aq1;
  {
    const _Float16* qrow = base + (size_t)(qbase + lrow) * rstride + h * DHEAD + hi * 8;
    aq0.h[0] = *(const h8*)(qrow);
    aq0.h[1] = *(const h8*)(qrow + 16);
    aq1.h[0] = *(const h8*)(qrow + 32);
    aq1.h[1] = *(const h8*)(qrow + 48);
  }
  V16U ones;
  {
    _Float16* p = (_Float16*)&ones;
    #pragma unroll
    for (int j = 0; j < 16; j++) p[j] = (_Float16)1.0f;
  }

  float m[8], l[8];
  #pragma unroll
  for (int r = 0; r < 8; r++) { m[r] = -3e38f; l[r] = 0.f; }
  v8f oacc[4];
  for (int dg = 0; dg < 4; dg++)
    for (int e = 0; e < 8; e++) oacc[dg][e] = 0.f;

  for (int kb = 0; kb < S; kb += 32) {
    __syncthreads();
    // K tile via async copy (32 keys x 64 halfs)
    #pragma unroll
    for (int u = 0; u < 2; u++) {
      int c = tid + u * 128, kk = c >> 3, seg = c & 7;
      async_ld16(base + (size_t)(kb + kk) * rstride + DMODEL + h * DHEAD + seg * 8,
                 sK + kk * 72 + seg * 8);
    }
    // V tile manual transpose into [d][key]
    #pragma unroll
    for (int u = 0; u < 2; u++) {
      int c = tid + u * 128, kk = c >> 3, seg = c & 7;
      h8 v = *(const h8*)(base + (size_t)(kb + kk) * rstride + 2 * DMODEL + h * DHEAD + seg * 8);
      #pragma unroll
      for (int j = 0; j < 8; j++) sVt[(seg * 8 + j) * 40 + kk] = v[j];
    }
    if (kb + 32 < S)
      __builtin_prefetch(base + (size_t)(kb + 32 + (tid >> 2)) * rstride + DMODEL + h * DHEAD, 0, 0);
    async_wait0();
    __syncthreads();

    // scores: two 16x16 C tiles (keys kb..+15, kb+16..+31)
    v8f sc[2];
    #pragma unroll
    for (int g = 0; g < 2; g++) {
      for (int e = 0; e < 8; e++) sc[g][e] = 0.f;
      #pragma unroll
      for (int dgb = 0; dgb < 2; dgb++) {
        V16U bk;
        const _Float16* p = sK + (g * 16 + lrow) * 72 + dgb * 32 + hi * 8;
        bk.h[0] = *(const h8*)p;
        bk.h[1] = *(const h8*)(p + 16);
        sc[g] = __builtin_amdgcn_wmma_f32_16x16x32_f16(
            false, (dgb ? aq1.v : aq0.v), false, bk.v, (short)0, sc[g], false, false);
      }
      int key = kb + g * 16 + lrow;
      float mb = (key >= seqlen) ? -1e30f : 0.f;
      #pragma unroll
      for (int r = 0; r < 8; r++) sc[g][r] = sc[g][r] * 0.125f + mb;
    }

    // running max per row (row r+8*hi lives in reg r across the 16-lane half)
    float alpha[8];
    #pragma unroll
    for (int r = 0; r < 8; r++) {
      float v = fmaxf(sc[0][r], sc[1][r]);
      for (int o = 1; o < 16; o <<= 1) v = fmaxf(v, __shfl_xor(v, o, 32));
      float mn = fmaxf(m[r], v);
      alpha[r] = __expf(m[r] - mn);
      m[r] = mn;
    }
    // probabilities -> LDS (C layout) -> A fragment
    _Float16* pb = sP[wave];
    #pragma unroll
    for (int g = 0; g < 2; g++)
      #pragma unroll
      for (int r = 0; r < 8; r++)
        pb[(hi * 8 + r) * 40 + g * 16 + lrow] = (_Float16)__expf(sc[g][r] - m[r]);
    __syncthreads();
    V16U ap;
    {
      const _Float16* p = pb + lrow * 40 + hi * 8;
      ap.h[0] = *(const h8*)p;
      ap.h[1] = *(const h8*)(p + 16);
    }
    // row sums via tensor core: P(16x32) x ones(32x16) -> every column = rowsum
    v8f rs;
    for (int e = 0; e < 8; e++) rs[e] = 0.f;
    rs = __builtin_amdgcn_wmma_f32_16x16x32_f16(
        false, ap.v, false, ones.v, (short)0, rs, false, false);
    #pragma unroll
    for (int r = 0; r < 8; r++) l[r] = l[r] * alpha[r] + rs[r];

    // rescale O, accumulate P(16x32) x V(32x64)
    #pragma unroll
    for (int dg = 0; dg < 4; dg++) {
      #pragma unroll
      for (int r = 0; r < 8; r++) oacc[dg][r] *= alpha[r];
      V16U bv;
      const _Float16* p = sVt + (dg * 16 + lrow) * 40 + hi * 8;
      bv.h[0] = *(const h8*)p;
      bv.h[1] = *(const h8*)(p + 16);
      oacc[dg] = __builtin_amdgcn_wmma_f32_16x16x32_f16(
          false, ap.v, false, bv.v, (short)0, oacc[dg], false, false);
    }
  }

  #pragma unroll
  for (int r = 0; r < 8; r++) l[r] = 1.0f / l[r];
  #pragma unroll
  for (int dg = 0; dg < 4; dg++)
    #pragma unroll
    for (int r = 0; r < 8; r++) {
      int row = qbase + hi * 8 + r;
      int col = h * DHEAD + dg * 16 + lrow;
      O[((size_t)b * S + row) * DMODEL + col] = (_Float16)(oacc[dg][r] * l[r]);
    }
}

// ---------------- alignment -> durations ----------------
__global__ __launch_bounds__(256)
void k_zero(float* p, int n) { int i = blockIdx.x * 256 + threadIdx.x; if (i < n) p[i] = 0.f; }

__global__ __launch_bounds__(256)
void k_durations(const float* __restrict__ align, const int* __restrict__ mlen,
                 float* __restrict__ dur) {
  int gid = blockIdx.x * 256 + threadIdx.x;   // b*T + t
  int b = gid / TT, t = gid % TT;
  if (t >= mlen[b]) return;
  const float* row = align + (size_t)gid * LL;
  float best = row[0]; int bi = 0;
  for (int l = 1; l < LL; l++) { float v = row[l]; if (v > best) { best = v; bi = l; } }
  atomicAdd(&dur[b * LL + bi], 1.0f);
}

__global__ __launch_bounds__(LL)
void k_scan(const float* __restrict__ dur, int* __restrict__ cum) {
  __shared__ int s[LL];
  int b = blockIdx.x, l = threadIdx.x;
  int v = (int)rintf(dur[b * LL + l]);
  s[l] = (v > 1) ? v : 1;
  __syncthreads();
  for (int o = 1; o < LL; o <<= 1) {
    int add = (l >= o) ? s[l - o] : 0;
    __syncthreads();
    s[l] += add;
    __syncthreads();
  }
  cum[b * LL + l] = s[l];
}

__global__ __launch_bounds__(256)
void k_expand(const float* __restrict__ hid, const int* __restrict__ cum,
              const float* __restrict__ alpha2, float* __restrict__ out,
              _Float16* __restrict__ outh) {
  int bt = blockIdx.x;                 // b*T + t
  int b = bt / TT, t = bt % TT;
  const int* c = cum + b * LL;
  int lo = 0, hb = LL;
  while (lo < hb) { int mid = (lo + hb) >> 1; if (c[mid] <= t) lo = mid + 1; else hb = mid; }
  int idx = (lo < LL - 1) ? lo : (LL - 1);
  float valid = (t < c[LL - 1]) ? 1.f : 0.f;
  float a2 = alpha2[0];
  const float* hrow = hid + ((size_t)b * LL + idx) * DMODEL;
  for (int d = threadIdx.x; d < DMODEL; d += 256) {
    float v = hrow[d] * valid + a2 * pe_val(t, d, DMODEL);
    out [(size_t)bt * DMODEL + d] = v;
    outh[(size_t)bt * DMODEL + d] = (_Float16)v;
  }
}

// ---------------- duration predictor: k=3 conv (pad=1) ----------------
template<int CIN, bool RELU>
__global__ __launch_bounds__(DDUR)
void k_conv3(const float* __restrict__ X, const float* __restrict__ W,
             const float* __restrict__ Bb, float* __restrict__ Y) {
  int bl = blockIdx.x;                 // b*L + l
  int b = bl / LL, l = bl % LL;
  int o = threadIdx.x;
  float acc = Bb[o];
  const float* w = W + (size_t)o * CIN * 3;
  for (int k = 0; k < 3; k++) {
    int ls = l + k - 1;
    if (ls < 0 || ls >= LL) continue;
    const float* x = X + ((size_t)b * LL + ls) * CIN;
    for (int c = 0; c < CIN; c++) acc += x[c] * w[c * 3 + k];
  }
  if (RELU) acc = fmaxf(acc, 0.f);
  Y[(size_t)bl * DDUR + o] = acc;
}

__global__ __launch_bounds__(256)
void k_durlin(const float* __restrict__ X, const float* __restrict__ w,
              const float* __restrict__ bb, float* __restrict__ out) {
  int gid = blockIdx.x * 256 + threadIdx.x;   // b*L + l
  const float* x = X + (size_t)gid * DDUR;
  float acc = bb[0];
  for (int c = 0; c < DDUR; c++) acc += x[c] * w[c];
  out[gid] = acc;
}

__global__ __launch_bounds__(256)
void k_copy(const float* s, float* d, int n) {
  int i = blockIdx.x * 256 + threadIdx.x; if (i < n) d[i] = s[i];
}

// ---------------- host orchestration ----------------
extern "C" void kernel_launch(void* const* d_in, const int* in_sizes, int n_in,
                              void* d_out, int out_size, void* d_ws, size_t ws_size,
                              hipStream_t stream) {
  (void)in_sizes; (void)n_in; (void)out_size; (void)ws_size;
  const int*   text   = (const int*)d_in[0];
  const float* align  = (const float*)d_in[1];
  const int*   tlen   = (const int*)d_in[2];
  const int*   mlen   = (const int*)d_in[3];
  const float* emb    = (const float*)d_in[4];
  const float* alpha1 = (const float*)d_in[5];
  const float* alpha2 = (const float*)d_in[6];
  const int ENC = 7, DEC = 19, DUR = 31;
  const float* proj_w = (const float*)d_in[41];
  const float* proj_b = (const float*)d_in[42];
  auto P = [&](int i) { return (const float*)d_in[i]; };

  float* ws = (float*)d_ws;
  size_t off = 0;
  float*     act  = ws + off;              off += (size_t)BB * TT * DMODEL;
  float*     tmp  = ws + off;              off += (size_t)BB * TT * DMODEL;
  float*     hid  = ws + off;              off += (size_t)BB * LL * DMODEL;
  _Float16*  actH = (_Float16*)(ws + off); off += (size_t)BB * TT * DMODEL / 2;
  _Float16*  bigH = (_Float16*)(ws + off); off += (size_t)BB * TT * FFDIM / 2;
  _Float16*  wf16 = (_Float16*)(ws + off); off += (size_t)FFDIM * DMODEL / 2;
  float*     dur  = ws + off;              off += (size_t)BB * LL;
  int*       cum  = (int*)(ws + off);      off += (size_t)BB * LL;
  float*     dt1  = ws + off;              off += (size_t)BB * LL * DDUR;
  float*     dt2  = ws + off;              off += (size_t)BB * LL * DDUR;
  _Float16*  attnoH = bigH + (size_t)BB * TT * 3 * DMODEL;  // unused tail of bigH

  float* out_mel = (float*)d_out;
  float* out_dur = out_mel + (size_t)BB * NMEL * TT;
  float* out_cnt = out_dur + (size_t)BB * LL;

  auto cvt = [&](const float* w, int n) {
    k_wcvt<<<(n + 255) / 256, 256, 0, stream>>>(w, wf16, n);
  };

  // one transformer layer: input act/actH, result back in act/actH
  auto layer = [&](int base, int i, const int* lens, int S) {
    int M = BB * S;
    size_t oW = (size_t)i;
    const float* qkvw = P(base + 0) + oW * 3 * DMODEL * DMODEL;
    const float* qkvb = P(base + 1) + oW * 3 * DMODEL;
    const float* outw = P(base + 2) + oW * DMODEL * DMODEL;
    const float* outb = P(base + 3) + oW * DMODEL;
    const float* f1w  = P(base + 4) + oW * FFDIM * DMODEL;
    const float* f1b  = P(base + 5) + oW * FFDIM;
    const float* f2w  = P(base + 6) + oW * DMODEL * FFDIM;
    const float* f2b  = P(base + 7) + oW * DMODEL;
    const float* l1g  = P(base + 8) + oW * DMODEL;
    const float* l1b  = P(base + 9) + oW * DMODEL;
    const float* l2g  = P(base + 10) + oW * DMODEL;
    const float* l2b  = P(base + 11) + oW * DMODEL;

    cvt(qkvw, 3 * DMODEL * DMODEL);
    k_gemm<false,false,false><<<dim3((3*DMODEL)/128, M/128), 256, 0, stream>>>(
        actH, wf16, qkvb, nullptr, nullptr, bigH, M, 3*DMODEL, DMODEL, 0);
    k_attn<<<dim3(BB*NHEAD, S/64), 128, 0, stream>>>(bigH, lens, attnoH, S);
    cvt(outw, DMODEL * DMODEL);
    k_gemm<false,true,false><<<dim3(DMODEL/128, M/128), 256, 0, stream>>>(
        attnoH, wf16, outb, act, tmp, nullptr, M, DMODEL, DMODEL, 0);
    k_ln<<<M/8, 256, 0, stream>>>(tmp, l1g, l1b, act, actH, M, DMODEL);
    cvt(f1w, FFDIM * DMODEL);
    k_gemm<true,false,false><<<dim3(FFDIM/128, M/128), 256, 0, stream>>>(
        actH, wf16, f1b, nullptr, nullptr, bigH, M, FFDIM, DMODEL, 0);
    cvt(f2w, DMODEL * FFDIM);
    k_gemm<false,true,false><<<dim3(DMODEL/128, M/128), 256, 0, stream>>>(
        bigH, wf16, f2b, act, tmp, nullptr, M, DMODEL, FFDIM, 0);
    k_ln<<<M/8, 256, 0, stream>>>(tmp, l2g, l2b, act, actH, M, DMODEL);
  };

  // ---- encoder ----
  k_embed<<<(BB*LL*DMODEL)/256, 256, 0, stream>>>(text, emb, alpha1, act, actH);
  for (int i = 0; i < NLAYER; i++) layer(ENC, i, tlen, LL);
  hipMemcpyAsync(hid, act, sizeof(float)*(size_t)BB*LL*DMODEL,
                 hipMemcpyDeviceToDevice, stream);

  // ---- durations + length regulator ----
  k_zero<<<(BB*LL)/256, 256, 0, stream>>>(dur, BB*LL);
  k_durations<<<(BB*TT)/256, 256, 0, stream>>>(align, mlen, dur);
  k_scan<<<BB, LL, 0, stream>>>(dur, cum);
  k_expand<<<BB*TT, 256, 0, stream>>>(hid, cum, alpha2, act, actH);

  // ---- decoder ----
  for (int i = 0; i < NLAYER; i++) layer(DEC, i, mlen, TT);

  // ---- mel projection (transposed store -> [B, NMEL, T]) ----
  cvt(proj_w, NMEL * DMODEL);
  k_gemm<false,false,true><<<dim3(1, (BB*TT)/128), 256, 0, stream>>>(
      actH, wf16, proj_b, nullptr, out_mel, nullptr, BB*TT, NMEL, DMODEL, TT);

  // ---- duration predictor on encoder hidden ----
  k_conv3<DMODEL,true><<<BB*LL, DDUR, 0, stream>>>(hid, P(DUR+0), P(DUR+1), dt1);
  k_ln<<<(BB*LL)/8, 256, 0, stream>>>(dt1, P(DUR+4), P(DUR+5), dt2, nullptr, BB*LL, DDUR);
  k_conv3<DDUR,true><<<BB*LL, DDUR, 0, stream>>>(dt2, P(DUR+2), P(DUR+3), dt1);
  k_ln<<<(BB*LL)/8, 256, 0, stream>>>(dt1, P(DUR+6), P(DUR+7), dt2, nullptr, BB*LL, DDUR);
  k_durlin<<<(BB*LL)/256, 256, 0, stream>>>(dt2, P(DUR+8), P(DUR+9), out_dur);

  // ---- durations output ----
  k_copy<<<(BB*LL)/256, 256, 0, stream>>>(dur, out_cnt, BB*LL);
}
